// CausalSelfAttention_51110110822880
// MI455X (gfx1250) — matrix operand order repaired and checked by reading, not compiled, not run on hardware
//
#include <hip/hip_runtime.h>
#include <hip/hip_bf16.h>

// ---------------------------------------------------------------------------
// CausalSelfAttention for MI455X (gfx1250, wave32, WMMA).
// bf16 inputs to v_wmma_f32_16x16x32_bf16, fp32 accumulation throughout.
// Round 3: GEMM tile staging switched to CDNA5 async global->LDS DMA
// (global_load_async_to_lds_b128 / s_wait_asynccnt), saving the VGPR
// round-trip and overlapping tile fetch with WMMA under ASYNCcnt.
// ---------------------------------------------------------------------------

typedef unsigned short u16t;
typedef __bf16 bf16_t;
typedef bf16_t v16bf __attribute__((ext_vector_type(16)));
typedef float v8f __attribute__((ext_vector_type(8)));
typedef unsigned int u32x4 __attribute__((ext_vector_type(4)));

union Frag {
    v16bf v;
    u32x4 q[2];
};

__device__ __forceinline__ u16t f2bf(float f) {
    unsigned int u = __float_as_uint(f);
    u += 0x7fffu + ((u >> 16) & 1u);   // round-to-nearest-even
    return (u16t)(u >> 16);
}

// A-matrix fragment (16x32 bf16, M=row per lane&15):
// lane-half h: elems 0..7  <- K = 8h .. 8h+7 ; elems 8..15 <- K = 16+8h .. +7
__device__ __forceinline__ v16bf load_a_frag(const u16t* p, int half) {
    Frag f;
    f.q[0] = *(const u32x4*)(p + half * 8);
    f.q[1] = *(const u32x4*)(p + 16 + half * 8);
    return f.v;
}

// B-matrix fragment (32x16 bf16, N=col per lane&15):
// lane-half h: elems 0..7 <- K = 8h .. 8h+7 ; elems 8..15 <- K = 16+8h .. +7
__device__ __forceinline__ v16bf load_b_frag(const u16t* p, int half) {
    Frag f;
    f.q[0] = *(const u32x4*)(p + half * 8);
    f.q[1] = *(const u32x4*)(p + 16 + half * 8);
    return f.v;
}

__device__ __forceinline__ v8f wmma_bf16(v16bf a, v16bf b, v8f c) {
    return __builtin_amdgcn_wmma_f32_16x16x32_bf16(
        false, a, false, b, (short)0, c, false, false);
}

// ---------------------------------------------------------------------------
// CDNA5 async global -> LDS copy (ASYNCcnt path). Each lane moves 16 bytes:
// LDS[lds_off] = MEM[gaddr]. Inline asm so it works on both toolchains.
// ---------------------------------------------------------------------------
__device__ __forceinline__ void async_copy_b128(unsigned lds_off, const void* gptr) {
    asm volatile("global_load_async_to_lds_b128 %0, %1, off"
                 :: "v"(lds_off), "v"((unsigned long long)(size_t)gptr)
                 : "memory");
}
__device__ __forceinline__ void wait_async0() {
    asm volatile("s_wait_asynccnt 0x0" ::: "memory");
}
// Generic pointer to __shared__ carries the LDS byte offset in bits [31:0].
__device__ __forceinline__ unsigned lds_off32(const void* p) {
    return (unsigned)(size_t)p;
}

// ---------------------------------------------------------------------------
// DPP cross-lane reductions over the 16-lane half (wave32).
// xor1/2/4 via DPP8 lane-permute, xor8 via DPP16 ROW_MIRROR (valid because
// after xor1/2/4 every 8-lane group is uniform, so mirror == xor8).
// All VALU: no LDS traffic, no waits.
// ---------------------------------------------------------------------------
#define DPP8_SEL(a,b,c,d,e,f,g,h) \
    ((a)|((b)<<3)|((c)<<6)|((d)<<9)|((e)<<12)|((f)<<15)|((g)<<18)|((h)<<21))

template <int SEL>
__device__ __forceinline__ float dpp8f(float x) {
    return __int_as_float(__builtin_amdgcn_mov_dpp8(__float_as_int(x), SEL));
}
__device__ __forceinline__ float dpp_row_mirror(float x) {
    int xi = __float_as_int(x);
    return __int_as_float(__builtin_amdgcn_update_dpp(xi, xi, 0x140, 0xf, 0xf, true));
}
__device__ __forceinline__ float rowmax16(float x) {
    x = fmaxf(x, dpp8f<DPP8_SEL(1,0,3,2,5,4,7,6)>(x));
    x = fmaxf(x, dpp8f<DPP8_SEL(2,3,0,1,6,7,4,5)>(x));
    x = fmaxf(x, dpp8f<DPP8_SEL(4,5,6,7,0,1,2,3)>(x));
    x = fmaxf(x, dpp_row_mirror(x));
    return x;
}
__device__ __forceinline__ float rowsum16(float x) {
    x += dpp8f<DPP8_SEL(1,0,3,2,5,4,7,6)>(x);
    x += dpp8f<DPP8_SEL(2,3,0,1,6,7,4,5)>(x);
    x += dpp8f<DPP8_SEL(4,5,6,7,0,1,2,3)>(x);
    x += dpp_row_mirror(x);
    return x;
}

// ---------------------------------------------------------------------------
// Elementwise fp32 -> bf16 convert
// ---------------------------------------------------------------------------
__global__ __launch_bounds__(256) void f32_to_bf16_kernel(const float* __restrict__ in,
                                                          u16t* __restrict__ out,
                                                          long long n) {
    long long i = (long long)blockIdx.x * blockDim.x + threadIdx.x;
    if (i < n) out[i] = f2bf(in[i]);
}

// ---------------------------------------------------------------------------
// Tiled bf16 GEMM: C[M,N] = A[M,K] * W[N,K]^T   (torch Linear layout)
// Block 256 threads = 8 waves; tile 128x128x32; wave tile 64x32 (4x2 WMMA).
// Double-buffered LDS (40 KB), tiles staged with async global->LDS DMA so
// the next tile streams in under the current tile's 8 WMMAs.
// ---------------------------------------------------------------------------
template <int OUT_BF16>
__global__ __launch_bounds__(256) void gemm_bf16_kernel(const u16t* __restrict__ A,
                                                        const u16t* __restrict__ W,
                                                        float* __restrict__ Cf,
                                                        u16t* __restrict__ Cb,
                                                        int M, int N, int K) {
    __shared__ __align__(16) u16t As[2][128][40];   // 80B stride: conflict-free
    __shared__ __align__(16) u16t Bs[2][128][40];

    const int tid = threadIdx.x;
    const int wid = tid >> 5;
    const int lane = tid & 31;
    const int half = lane >> 4;
    const int id = lane & 15;
    const int wr = wid >> 2;      // 0..1
    const int wc = wid & 3;       // 0..3
    const int m0 = blockIdx.y * 128;
    const int n0 = blockIdx.x * 128;

    const int r0 = tid >> 2;          // 0..63 : first row this thread loads
    const int ch = (tid & 3) * 8;     // k-chunk offset (elements)

    v8f acc[4][2];
#pragma unroll
    for (int mi = 0; mi < 4; ++mi)
#pragma unroll
        for (int ni = 0; ni < 2; ++ni)
#pragma unroll
            for (int r = 0; r < 8; ++r) acc[mi][ni][r] = 0.0f;

    // async DMA of one 128x32 A tile + 128x32 B tile into buffer `buf`
    auto stage_tile = [&](int buf, int k0) {
        async_copy_b128(lds_off32(&As[buf][r0][ch]),
                        &A[(size_t)(m0 + r0) * K + k0 + ch]);
        async_copy_b128(lds_off32(&As[buf][r0 + 64][ch]),
                        &A[(size_t)(m0 + r0 + 64) * K + k0 + ch]);
        async_copy_b128(lds_off32(&Bs[buf][r0][ch]),
                        &W[(size_t)(n0 + r0) * K + k0 + ch]);
        async_copy_b128(lds_off32(&Bs[buf][r0 + 64][ch]),
                        &W[(size_t)(n0 + r0 + 64) * K + k0 + ch]);
    };

    stage_tile(0, 0);
    wait_async0();
    __syncthreads();

    const int nk = K >> 5;
    for (int kt = 0; kt < nk; ++kt) {
        const int buf = kt & 1;
        if (kt + 1 < nk) stage_tile(buf ^ 1, (kt + 1) << 5);   // DMA next tile

        v16bf af[4], bfg[2];
#pragma unroll
        for (int mi = 0; mi < 4; ++mi)
            af[mi] = load_a_frag(&As[buf][wr * 64 + mi * 16 + id][0], half);
#pragma unroll
        for (int ni = 0; ni < 2; ++ni)
            bfg[ni] = load_b_frag(&Bs[buf][wc * 32 + ni * 16 + id][0], half);

#pragma unroll
        for (int mi = 0; mi < 4; ++mi)
#pragma unroll
            for (int ni = 0; ni < 2; ++ni)
                acc[mi][ni] = wmma_bf16(af[mi], bfg[ni], acc[mi][ni]);

        if (kt + 1 < nk) wait_async0();          // next tile landed
        __syncthreads();
    }

    // epilogue: C/D layout — vgpr r holds row (r + 8*half), col = id
#pragma unroll
    for (int mi = 0; mi < 4; ++mi)
#pragma unroll
        for (int ni = 0; ni < 2; ++ni)
#pragma unroll
            for (int r = 0; r < 8; ++r) {
                int row = m0 + wr * 64 + mi * 16 + r + 8 * half;
                int col = n0 + wc * 32 + ni * 16 + id;
                float v = acc[mi][ni][r];
                if (OUT_BF16) Cb[(size_t)row * N + col] = f2bf(v);
                else          Cf[(size_t)row * N + col] = v;
            }
}

// ---------------------------------------------------------------------------
// Split qkv [B,T,3C] bf16 -> Q[B,H,T,Dh], K[B,H,T,Dh], Vt[B,H,Dh,T]
// ---------------------------------------------------------------------------
__global__ __launch_bounds__(256) void split_qkv_kernel(const u16t* __restrict__ qkv,
                                                        u16t* __restrict__ Qb,
                                                        u16t* __restrict__ Kb,
                                                        u16t* __restrict__ Vt) {
    const long long n = 4LL * 2048 * 1024;   // B*T*C
    long long i = (long long)blockIdx.x * blockDim.x + threadIdx.x;
    if (i >= n) return;
    int d = (int)(i & 63);
    int h = (int)((i >> 6) & 15);
    int t = (int)((i >> 10) & 2047);
    int b = (int)(i >> 21);
    size_t base = ((size_t)b * 2048 + t) * 3072 + h * 64 + d;
    u16t q = qkv[base];
    u16t k = qkv[base + 1024];
    u16t v = qkv[base + 2048];
    size_t bh = (size_t)b * 16 + h;
    Qb[(bh * 2048 + t) * 64 + d] = q;
    Kb[(bh * 2048 + t) * 64 + d] = k;
    Vt[(bh * 64 + d) * 2048 + t] = v;
}

// ---------------------------------------------------------------------------
// Flash attention: grid (B*H, T/64), 128 threads = 4 waves, each wave = 16 q rows.
// Key blocks of 32; online softmax with DPP reductions; P respilled via LDS
// into A-fragment layout. K and V fragments loaded up-front so the softmax
// VALU work hides their latency before the P*V WMMAs.
// ---------------------------------------------------------------------------
__global__ __launch_bounds__(128) void attn_kernel(const u16t* __restrict__ Qb,
                                                   const u16t* __restrict__ Kb,
                                                   const u16t* __restrict__ Vt,
                                                   u16t* __restrict__ Yb) {
    constexpr int T = 2048, Dh = 64, C = 1024;
    __shared__ __align__(16) u16t Ps[4][16][40];   // per-wave 16x32 P tile (padded)

    const int tid = threadIdx.x;
    const int wid = tid >> 5;
    const int lane = tid & 31;
    const int half = lane >> 4;
    const int id = lane & 15;
    const int bh = blockIdx.x;          // b*16 + h
    const int b = bh >> 4;
    const int h = bh & 15;
    const int q0 = blockIdx.y * 64 + wid * 16;

    const u16t* Qp = Qb + (size_t)bh * T * Dh;
    const u16t* Kp = Kb + (size_t)bh * T * Dh;
    const u16t* Vp = Vt + (size_t)bh * Dh * T;

    // Q fragments, register-resident for the whole sweep (Dh=64 -> 2 K-chunks)
    v16bf qf0 = load_a_frag(Qp + (size_t)(q0 + id) * Dh, half);
    v16bf qf1 = load_a_frag(Qp + (size_t)(q0 + id) * Dh + 32, half);

    float mrun[8], lrun[8];
    v8f o[4];
#pragma unroll
    for (int r = 0; r < 8; ++r) { mrun[r] = -1e30f; lrun[r] = 0.0f; }
#pragma unroll
    for (int c = 0; c < 4; ++c)
#pragma unroll
        for (int r = 0; r < 8; ++r) o[c][r] = 0.0f;

    const int nkb = (q0 + 16 + 31) >> 5;   // causal: keys up to q0+15
    for (int kb = 0; kb < nkb; ++kb) {
        const int k0 = kb * 32;

        // Load all K and V fragments for this block up-front.
        v16bf kf[2][2], vf[4];
#pragma unroll
        for (int nb = 0; nb < 2; ++nb) {
            const u16t* krow = Kp + (size_t)(k0 + nb * 16 + id) * Dh;
            kf[nb][0] = load_b_frag(krow, half);
            kf[nb][1] = load_b_frag(krow + 32, half);
        }
#pragma unroll
        for (int c = 0; c < 4; ++c)
            vf[c] = load_b_frag(Vp + (size_t)(c * 16 + id) * T + k0, half);

        // S = (Q * K^T) * 0.125 : two 16x16 tiles
        v8f s[2];
#pragma unroll
        for (int nb = 0; nb < 2; ++nb) {
#pragma unroll
            for (int r = 0; r < 8; ++r) s[nb][r] = 0.0f;
            s[nb] = wmma_bf16(qf0, kf[nb][0], s[nb]);
            s[nb] = wmma_bf16(qf1, kf[nb][1], s[nb]);
        }

        const bool needMask = (k0 + 31 > q0);   // uniform per wave

        // online softmax per row (row's 16 cols live across a 16-lane half)
#pragma unroll
        for (int r = 0; r < 8; ++r) {
            const int qrow = q0 + r + 8 * half;
            float v0 = s[0][r] * 0.125f;
            float v1 = s[1][r] * 0.125f;
            if (needMask) {
                if (k0 + id > qrow)      v0 = -1e30f;
                if (k0 + 16 + id > qrow) v1 = -1e30f;
            }
            float rm = rowmax16(fmaxf(v0, v1));
            float mnew = fmaxf(mrun[r], rm);
            float p0 = __expf(v0 - mnew);
            float p1 = __expf(v1 - mnew);
            float rs = rowsum16(p0 + p1);
            float alpha = __expf(mrun[r] - mnew);
            lrun[r] = lrun[r] * alpha + rs;
            mrun[r] = mnew;
#pragma unroll
            for (int c = 0; c < 4; ++c) o[c][r] *= alpha;

            // spill P (C/D layout) -> LDS so we can reload it in A layout
            Ps[wid][r + 8 * half][id] = f2bf(p0);
            Ps[wid][r + 8 * half][16 + id] = f2bf(p1);
        }

        // O += P * V  (per-wave LDS region; DS ops in-order within a wave)
        v16bf pf = load_a_frag(&Ps[wid][id][0], half);
#pragma unroll
        for (int c = 0; c < 4; ++c) o[c] = wmma_bf16(pf, vf[c], o[c]);
    }

    // normalize + scatter to y [B,T,C] bf16 (feeds out-proj GEMM)
#pragma unroll
    for (int r = 0; r < 8; ++r) {
        float inv = 1.0f / lrun[r];
        int row = b * T + q0 + r + 8 * half;
#pragma unroll
        for (int c = 0; c < 4; ++c) {
            int col = h * 64 + c * 16 + id;
            Yb[(size_t)row * C + col] = f2bf(o[c][r] * inv);
        }
    }
}

// ---------------------------------------------------------------------------
// Host-side orchestration
// ---------------------------------------------------------------------------
extern "C" void kernel_launch(void* const* d_in, const int* in_sizes, int n_in,
                              void* d_out, int out_size, void* d_ws, size_t ws_size,
                              hipStream_t stream) {
    constexpr int B = 4, T = 2048, C = 1024, H = 16;
    const float* x = (const float*)d_in[0];
    const float* w_qkv = (const float*)d_in[1];
    const float* w_out = (const float*)d_in[2];
    float* out = (float*)d_out;

    size_t off = 0;
    auto carve = [&](size_t bytes) -> void* {
        void* p = (char*)d_ws + off;
        off += (bytes + 255) & ~(size_t)255;
        return p;
    };
    u16t* xb   = (u16t*)carve((size_t)B * T * C * 2);       // 16 MB
    u16t* wqb  = (u16t*)carve((size_t)3 * C * C * 2);       // 6 MB
    u16t* wob  = (u16t*)carve((size_t)C * C * 2);           // 2 MB
    u16t* qkvb = (u16t*)carve((size_t)B * T * 3 * C * 2);   // 48 MB
    u16t* Qb   = (u16t*)carve((size_t)B * T * C * 2);       // 16 MB
    u16t* Kb   = (u16t*)carve((size_t)B * T * C * 2);       // 16 MB
    u16t* Vt   = (u16t*)carve((size_t)B * T * C * 2);       // 16 MB
    u16t* Yb   = (u16t*)carve((size_t)B * T * C * 2);       // 16 MB

    const long long nx = (long long)B * T * C;
    const long long nwq = 3LL * C * C;
    const long long nwo = (long long)C * C;
    f32_to_bf16_kernel<<<(unsigned)((nx  + 255) / 256), 256, 0, stream>>>(x, xb, nx);
    f32_to_bf16_kernel<<<(unsigned)((nwq + 255) / 256), 256, 0, stream>>>(w_qkv, wqb, nwq);
    f32_to_bf16_kernel<<<(unsigned)((nwo + 255) / 256), 256, 0, stream>>>(w_out, wob, nwo);

    // qkv = x @ w_qkv^T : [8192, 3072]
    gemm_bf16_kernel<1><<<dim3(3 * C / 128, B * T / 128), 256, 0, stream>>>(
        xb, wqb, nullptr, qkvb, B * T, 3 * C, C);

    split_qkv_kernel<<<(unsigned)((nx + 255) / 256), 256, 0, stream>>>(qkvb, Qb, Kb, Vt);

    attn_kernel<<<dim3(B * H, T / 64), 128, 0, stream>>>(Qb, Kb, Vt, Yb);

    // out = y @ w_out^T : [8192, 1024] fp32
    gemm_bf16_kernel<0><<<dim3(C / 128, B * T / 128), 256, 0, stream>>>(
        Yb, wob, out, nullptr, B * T, C, C);
}
